// GCNEncoder_34333968564540
// MI455X (gfx1250) — matrix operand order repaired and checked by reading, not compiled, not run on hardware
//
#include <hip/hip_runtime.h>
#include <hip/hip_bf16.h>
#include <stdint.h>

// ---------------------------------------------------------------------------
// GCN 2-layer forward for gfx1250 (MI455X).
//   h1 = X @ W1^T                (WMMA f16 -> f32)
//   a1 = Dinv (A+I) Dinv h1 + b1 (atomic scatter over edges + self-loop)
//   h  = dropout(relu(a1))       (Threefry2x32 replication of jax key(42))
//   h2 = h  @ W2^T               (WMMA f16 -> f32)
//   out= Dinv (A+I) Dinv h2 + b2
// ---------------------------------------------------------------------------

#define GCN_IN   256
#define GCN_HID  128

typedef __attribute__((ext_vector_type(16))) _Float16 v16h;
typedef __attribute__((ext_vector_type(8)))  _Float16 v8h;
typedef __attribute__((ext_vector_type(8)))  float    v8f;

__device__ __forceinline__ uint32_t rotl32(uint32_t x, uint32_t d) {
  return (x << d) | (x >> (32u - d));
}

// Replicates jax.random.bernoulli(jax.random.key(42), 0.9, shape) for flat
// element j of a size-S array (S even). Threefry2x32, key = {0, 42}.
__device__ __forceinline__ bool dropout_keep(uint32_t j, uint32_t S) {
  const uint32_t half = S >> 1;
  const bool first = j < half;
  uint32_t x0, x1;
  if (first) { x0 = j;        x1 = j + half; }
  else       { x0 = j - half; x1 = j;        }
  const uint32_t ks0 = 0u, ks1 = 42u, ks2 = 0x1BD11BDAu ^ 42u;
  x0 += ks0; x1 += ks1;
#define TF_R(r) { x0 += x1; x1 = rotl32(x1, r); x1 ^= x0; }
  TF_R(13) TF_R(15) TF_R(26) TF_R(6)
  x0 += ks1; x1 += ks2 + 1u;
  TF_R(17) TF_R(29) TF_R(16) TF_R(24)
  x0 += ks2; x1 += ks0 + 2u;
  TF_R(13) TF_R(15) TF_R(26) TF_R(6)
  x0 += ks0; x1 += ks1 + 3u;
  TF_R(17) TF_R(29) TF_R(16) TF_R(24)
  x0 += ks1; x1 += ks2 + 4u;
  TF_R(13) TF_R(15) TF_R(26) TF_R(6)
  x0 += ks2; x1 += ks0 + 5u;
#undef TF_R
  const uint32_t bits = first ? x0 : x1;
  const float u = __uint_as_float((bits >> 9) | 0x3f800000u) - 1.0f;
  return u < 0.9f;
}

// ---------------------------------------------------------------------------
// fp32 -> f16 weight conversion
__global__ void cvt_f16_kernel(const float* __restrict__ in,
                               _Float16* __restrict__ out, int n) {
  int i = blockIdx.x * blockDim.x + threadIdx.x;
  if (i < n) out[i] = (_Float16)in[i];
}

// Degree accumulation over dst (self-loops added analytically later).
__global__ void deg_kernel(const long long* __restrict__ dst,
                           float* __restrict__ deg, int E) {
  int i = blockIdx.x * blockDim.x + threadIdx.x;
  if (i < E)
    __hip_atomic_fetch_add(&deg[(int)dst[i]], 1.0f,
                           __ATOMIC_RELAXED, __HIP_MEMORY_SCOPE_AGENT);
}

// dinv[i] = rsqrt(deg[i] + 1)   (+1 = self loop; in-place over deg buffer)
__global__ void dinv_kernel(float* __restrict__ d, int n) {
  int i = blockIdx.x * blockDim.x + threadIdx.x;
  if (i < n) d[i] = rsqrtf(d[i] + 1.0f);
}

// ---------------------------------------------------------------------------
// GEMM: C[M,128] = A[M,K] @ B^T, B given row-major (128,K) as f16.
// One wave computes a 16-row x 128-col strip. K in {256,128}, multiple of 32.
// All 8 B fragments of a k-chunk are loaded up front (one clause, 16 b128
// loads in flight) so the 8-WMMA chain overlaps the remaining loads.
// A source is fp32 (AHALF=0) or f16 (AHALF=1).
template <int K, int AHALF>
__global__ void gemm_wmma_kernel(const void* __restrict__ Ap,
                                 const _Float16* __restrict__ Bh,
                                 float* __restrict__ Cp, int Mtiles) {
  const int wave = blockIdx.x * (blockDim.x >> 5) + (threadIdx.x >> 5);
  if (wave >= Mtiles) return;              // wave-uniform: EXEC stays all-ones
  const int lane = threadIdx.x & 31;
  const int rc   = lane & 15;              // A row / B-C column within tile
  const int hs   = lane >> 4;              // lane-half select
  const int m0   = wave << 4;

  v8f acc[8];
#pragma unroll
  for (int t = 0; t < 8; ++t) acc[t] = (v8f){0.f,0.f,0.f,0.f,0.f,0.f,0.f,0.f};

  for (int k0 = 0; k0 < K; k0 += 32) {
    // ---- A fragment: lane half hs holds K = k0+hs*8 + {0..7, 16..23}
    v16h a;
    if (AHALF) {
      const _Float16* ar = (const _Float16*)Ap + (size_t)(m0 + rc) * K + k0 + hs * 8;
      v8h lo = *(const v8h*)(ar);
      v8h hi = *(const v8h*)(ar + 16);
#pragma unroll
      for (int i = 0; i < 8; ++i) { a[i] = lo[i]; a[8 + i] = hi[i]; }
    } else {
      const float* ar = (const float*)Ap + (size_t)(m0 + rc) * K + k0 + hs * 8;
      float4 l0 = *(const float4*)(ar);
      float4 l1 = *(const float4*)(ar + 4);
      float4 h0 = *(const float4*)(ar + 16);
      float4 h1 = *(const float4*)(ar + 20);
      a[0]=(_Float16)l0.x; a[1]=(_Float16)l0.y; a[2]=(_Float16)l0.z; a[3]=(_Float16)l0.w;
      a[4]=(_Float16)l1.x; a[5]=(_Float16)l1.y; a[6]=(_Float16)l1.z; a[7]=(_Float16)l1.w;
      a[8]=(_Float16)h0.x; a[9]=(_Float16)h0.y; a[10]=(_Float16)h0.z; a[11]=(_Float16)h0.w;
      a[12]=(_Float16)h1.x; a[13]=(_Float16)h1.y; a[14]=(_Float16)h1.z; a[15]=(_Float16)h1.w;
    }

    // ---- issue ALL B-fragment loads for this k-chunk first
    // B[k][n] = W[n][k]; lane half hs holds K = k0 + hs*8 + {0..7, 16..23}
    v16h b[8];
#pragma unroll
    for (int nt = 0; nt < 8; ++nt) {
      const _Float16* br = Bh + (size_t)(nt * 16 + rc) * K + k0 + hs * 8;
      v8h b0 = *(const v8h*)(br);
      v8h b1 = *(const v8h*)(br + 16);
#pragma unroll
      for (int i = 0; i < 8; ++i) { b[nt][i] = b0[i]; b[nt][8 + i] = b1[i]; }
    }

    // ---- 8 WMMAs; waits on the in-flight loads decrement gradually
#pragma unroll
    for (int nt = 0; nt < 8; ++nt) {
      acc[nt] = __builtin_amdgcn_wmma_f32_16x16x32_f16(
          false, a, false, b[nt], (short)0, acc[nt], false, false);
    }
  }

  // ---- store: VGPR r -> (M = r + hs*8, N = rc) per 16x16 f32 C/D layout
#pragma unroll
  for (int nt = 0; nt < 8; ++nt) {
#pragma unroll
    for (int r = 0; r < 8; ++r) {
      const int mrow = m0 + r + hs * 8;
      Cp[(size_t)mrow * GCN_HID + nt * 16 + rc] = acc[nt][r];
    }
  }
}

// ---------------------------------------------------------------------------
// Edge scatter: acc[dst] += H[src] * dinv[src]*dinv[dst].
// One wave per edge; lane covers 4 of the 128 channels (float4 gather).
__global__ void scatter_kernel(const float* __restrict__ H,
                               const long long* __restrict__ src,
                               const long long* __restrict__ dst,
                               const float* __restrict__ dinv,
                               float* __restrict__ acc, int E) {
  const int gid  = blockIdx.x * blockDim.x + threadIdx.x;
  const int edge = gid >> 5;
  const int lane = gid & 31;
  if (edge >= E) return;
  const int s = (int)src[edge];
  const int d = (int)dst[edge];
  const float norm = dinv[s] * dinv[d];
  const float4 h4 = *(const float4*)(H + (size_t)s * GCN_HID + lane * 4);
  float* o = acc + (size_t)d * GCN_HID + lane * 4;
  __hip_atomic_fetch_add(o + 0, h4.x * norm, __ATOMIC_RELAXED, __HIP_MEMORY_SCOPE_AGENT);
  __hip_atomic_fetch_add(o + 1, h4.y * norm, __ATOMIC_RELAXED, __HIP_MEMORY_SCOPE_AGENT);
  __hip_atomic_fetch_add(o + 2, h4.z * norm, __ATOMIC_RELAXED, __HIP_MEMORY_SCOPE_AGENT);
  __hip_atomic_fetch_add(o + 3, h4.w * norm, __ATOMIC_RELAXED, __HIP_MEMORY_SCOPE_AGENT);
}

// Layer-1 finalize: + self-loop + bias, relu, dropout, emit f16 for GEMM2.
__global__ void finalize1_kernel(const float* __restrict__ acc,
                                 const float* __restrict__ H1,
                                 const float* __restrict__ dinv,
                                 const float* __restrict__ b1,
                                 _Float16* __restrict__ Hdrop, int total) {
  const int idx = blockIdx.x * blockDim.x + threadIdx.x;
  if (idx >= total) return;
  const int row = idx >> 7;          // / 128
  const int c   = idx & 127;
  const float di = dinv[row];
  float v = acc[idx] + H1[idx] * di * di + b1[c];
  v = fmaxf(v, 0.0f);
  v = dropout_keep((uint32_t)idx, (uint32_t)total) ? v * (1.0f / 0.9f) : 0.0f;
  Hdrop[idx] = (_Float16)v;
}

// Layer-2 finalize: out += self-loop + bias (out already holds edge sums).
__global__ void finalize2_kernel(float* __restrict__ out,
                                 const float* __restrict__ H2,
                                 const float* __restrict__ dinv,
                                 const float* __restrict__ b2, int total) {
  const int idx = blockIdx.x * blockDim.x + threadIdx.x;
  if (idx >= total) return;
  const int row = idx >> 7;
  const int c   = idx & 127;
  const float di = dinv[row];
  out[idx] = out[idx] + H2[idx] * di * di + b2[c];
}

// ---------------------------------------------------------------------------
extern "C" void kernel_launch(void* const* d_in, const int* in_sizes, int n_in,
                              void* d_out, int out_size, void* d_ws, size_t ws_size,
                              hipStream_t stream) {
  const float*      X    = (const float*)d_in[0];
  const long long*  eidx = (const long long*)d_in[1];   // int64 (2, E)
  const float*      W1   = (const float*)d_in[2];
  const float*      b1   = (const float*)d_in[3];
  const float*      W2   = (const float*)d_in[4];
  const float*      b2   = (const float*)d_in[5];
  float*            out  = (float*)d_out;

  const int N = in_sizes[0] / GCN_IN;        // 100000
  const int E = in_sizes[1] / 2;             // 1600000
  const long long* srcp = eidx;
  const long long* dstp = eidx + E;

  // workspace layout (16B-aligned slices)
  const size_t HB = (size_t)N * GCN_HID * sizeof(float);   // 51.2 MB
  char* ws = (char*)d_ws;
  float*    H     = (float*)(ws);                          // H1 then H2
  float*    acc1  = (float*)(ws + HB);
  _Float16* Hdrop = (_Float16*)(ws + 2 * HB);
  float*    dinv  = (float*)(ws + 2 * HB + HB / 2);        // also deg
  _Float16* W1h   = (_Float16*)(ws + 2 * HB + HB / 2 + ((size_t)N * 4 + 255) / 256 * 256);
  _Float16* W2h   = W1h + GCN_HID * GCN_IN;

  const int total  = N * GCN_HID;
  const int B256   = 256;
  const int Mtiles = N / 16;                 // 6250 (exact)

  // 0. weights -> f16
  cvt_f16_kernel<<<(GCN_HID * GCN_IN + B256 - 1) / B256, B256, 0, stream>>>(W1, W1h, GCN_HID * GCN_IN);
  cvt_f16_kernel<<<(GCN_HID * GCN_HID + B256 - 1) / B256, B256, 0, stream>>>(W2, W2h, GCN_HID * GCN_HID);

  // 1. degrees -> dinv
  hipMemsetAsync(dinv, 0, (size_t)N * sizeof(float), stream);
  deg_kernel<<<(E + B256 - 1) / B256, B256, 0, stream>>>(dstp, dinv, E);
  dinv_kernel<<<(N + B256 - 1) / B256, B256, 0, stream>>>(dinv, N);

  // 2. H1 = X @ W1^T   (WMMA)
  gemm_wmma_kernel<GCN_IN, 0><<<(Mtiles + 7) / 8, B256, 0, stream>>>(X, W1h, H, Mtiles);

  // 3. aggregate layer 1
  hipMemsetAsync(acc1, 0, HB, stream);
  {
    const int threads = E * 32;
    scatter_kernel<<<(threads + B256 - 1) / B256, B256, 0, stream>>>(H, srcp, dstp, dinv, acc1, E);
  }
  finalize1_kernel<<<(total + B256 - 1) / B256, B256, 0, stream>>>(acc1, H, dinv, b1, Hdrop, total);

  // 4. H2 = Hdrop @ W2^T   (WMMA, f16 input), reuse H buffer
  gemm_wmma_kernel<GCN_HID, 1><<<(Mtiles + 7) / 8, B256, 0, stream>>>(Hdrop, W2h, H, Mtiles);

  // 5. aggregate layer 2 directly into d_out
  hipMemsetAsync(out, 0, HB, stream);
  {
    const int threads = E * 32;
    scatter_kernel<<<(threads + B256 - 1) / B256, B256, 0, stream>>>(H, srcp, dstp, dinv, out, E);
  }
  finalize2_kernel<<<(total + B256 - 1) / B256, B256, 0, stream>>>(out, H, dinv, b2, total);
}
